// Attention_84842783965388
// MI455X (gfx1250) — matrix operand order repaired and checked by reading, not compiled, not run on hardware
//
#include <hip/hip_runtime.h>
#include <hip/hip_bf16.h>

#define B_ 64
#define S_ 2048
#define H_ 512
#define KP 40   // padded LDS K-stride in bf16 elems (80 B) -> conflict-free reads

typedef __attribute__((ext_vector_type(16))) __bf16 v16bf;
typedef __attribute__((ext_vector_type(8)))  __bf16 v8bf;
typedef __attribute__((ext_vector_type(4)))  __bf16 v4bf;
typedef __attribute__((ext_vector_type(8)))  float  v8f;

union V16 { v16bf v; v8bf h[2]; };

// ---------------------------------------------------------------------------
// Split f32 weights into bf16 hi/lo pair (bf16x2 decomposition).
// ---------------------------------------------------------------------------
__global__ __launch_bounds__(256)
void k_convert(const float* __restrict__ W, __bf16* __restrict__ hi,
               __bf16* __restrict__ lo) {
    int i = blockIdx.x * 256 + threadIdx.x;
    float x = W[i];
    __bf16 h = (__bf16)x;
    hi[i] = h;
    lo[i] = (__bf16)(x - (float)h);
}

// ---------------------------------------------------------------------------
// Wave-per-row GEMV: out[b,h] = dot(W[h,:], x[b,:]) + bias[h]
// Used for u1 = Wq @ tgt + bq  and  d' = Wref @ wsum + bref.
// ---------------------------------------------------------------------------
__global__ __launch_bounds__(256)
void k_gemv(const float* __restrict__ W, const float* __restrict__ x,
            const float* __restrict__ bias, float* __restrict__ out) {
    int gw   = blockIdx.x * 8 + (threadIdx.x >> 5);
    int lane = threadIdx.x & 31;
    int b = gw / H_, h = gw % H_;
    const float4* wr = (const float4*)(W + (size_t)h * H_);
    const float4* xr = (const float4*)(x + (size_t)b * H_);
    float acc = 0.0f;
#pragma unroll
    for (int c = 0; c < 4; ++c) {
        float4 wv = wr[c * 32 + lane];
        float4 xv = xr[c * 32 + lane];
        acc += wv.x * xv.x + wv.y * xv.y + wv.z * xv.z + wv.w * xv.w;
    }
#pragma unroll
    for (int off = 16; off; off >>= 1) acc += __shfl_xor(acc, off, 32);
    if (lane == 0) out[(size_t)b * H_ + h] = acc + bias[h];
}

// ---------------------------------------------------------------------------
// Main fused kernel: u2 tile GEMM (bf16x2 WMMA) + tanh/V reduction -> logit.
// 8 waves; wave w owns rows [64w,64w+64) x 32 s-columns.
// B tile (src 32k x 32s) converted ONCE per workgroup into double-buffered
// LDS (hi+lo), consumed via ds_load_b128 in WMMA B-fragment layout.
// ---------------------------------------------------------------------------
__global__ __launch_bounds__(256)
void k_main(const float* __restrict__ src,      // (B,S,H)
            const __bf16* __restrict__ Whi,     // (H,H) bf16 hi of Wref_w
            const __bf16* __restrict__ Wlo,     // (H,H) bf16 lo of Wref_w
            const float* __restrict__ Wref_b,   // (H)
            const float* __restrict__ Vv,       // (H)
            const float* __restrict__ u1,       // (B,H)
            float* __restrict__ logit)          // (B,S)
{
    const int b    = blockIdx.y;
    const int s0   = blockIdx.x * 32;
    const int tid  = threadIdx.x;
    const int wv   = tid >> 5;
    const int lane = tid & 31;
    const int hl   = lane >> 4;    // half-wave select
    const int l15  = lane & 15;
    const int mbase = wv * 64;

    __shared__ __align__(16) __bf16 bhL[2][32][KP];
    __shared__ __align__(16) __bf16 blL[2][32][KP];
    __shared__ float red[32];
    if (tid < 32) red[tid] = 0.0f;

    // Producer mapping: thread -> (column ps, k-chunk pk) of the 32x32 B tile.
    const int ps = tid >> 3;          // 0..31 : s-column
    const int pk = (tid & 7) * 4;     // 0,4,...,28 : k within chunk
    const float* psrc = src + ((size_t)b * S_ + (size_t)(s0 + ps)) * H_ + pk;

    auto produce = [&](int pbuf, int k0) {
        float4 f = *(const float4*)(psrc + k0);
        float xv[4] = {f.x, f.y, f.z, f.w};
        v4bf hv, lv;
#pragma unroll
        for (int i = 0; i < 4; ++i) {
            float x = xv[i];
            __bf16 h = (__bf16)x;
            hv[i] = h;
            lv[i] = (__bf16)(x - (float)h);
        }
        *(v4bf*)&bhL[pbuf][ps][pk] = hv;
        *(v4bf*)&blL[pbuf][ps][pk] = lv;
    };

    const v8f vzero = {0.f, 0.f, 0.f, 0.f, 0.f, 0.f, 0.f, 0.f};
    v8f acc[4][2];
#pragma unroll
    for (int t = 0; t < 4; ++t)
#pragma unroll
        for (int n = 0; n < 2; ++n) acc[t][n] = vzero;

    // A-matrix per-lane base offsets (row M = mbase + t*16 + l15).
    size_t aoff[4];
#pragma unroll
    for (int t = 0; t < 4; ++t)
        aoff[t] = (size_t)(mbase + t * 16 + l15) * H_ + hl * 8;

    produce(0, 0);
    __syncthreads();

    int buf = 0;
    for (int k0 = 0; k0 < H_; k0 += 32) {
        if (k0 + 32 < H_) produce(buf ^ 1, k0 + 32);     // stage next tile
        if (k0 + 64 < H_) __builtin_prefetch(psrc + k0 + 64, 0, 1);

        // B fragments from LDS: lane = col (n*16+l15), K = hl*8 .. hl*8+7, 16+hl*8 .. 16+hl*8+7.
        V16 bh[2], bl[2];
#pragma unroll
        for (int n = 0; n < 2; ++n) {
            int c = n * 16 + l15;
            bh[n].h[0] = *(const v8bf*)&bhL[buf][c][hl * 8];
            bh[n].h[1] = *(const v8bf*)&bhL[buf][c][hl * 8 + 16];
            bl[n].h[0] = *(const v8bf*)&blL[buf][c][hl * 8];
            bl[n].h[1] = *(const v8bf*)&blL[buf][c][hl * 8 + 16];
        }
#pragma unroll
        for (int t = 0; t < 4; ++t) {
            V16 ah, al;
            ah.h[0] = *(const v8bf*)(Whi + aoff[t] + k0);
            ah.h[1] = *(const v8bf*)(Whi + aoff[t] + k0 + 16);
            al.h[0] = *(const v8bf*)(Wlo + aoff[t] + k0);
            al.h[1] = *(const v8bf*)(Wlo + aoff[t] + k0 + 16);
#pragma unroll
            for (int n = 0; n < 2; ++n) {
                acc[t][n] = __builtin_amdgcn_wmma_f32_16x16x32_bf16(
                    false, ah.v, false, bh[n].v, (short)0, acc[t][n], false, false);
                acc[t][n] = __builtin_amdgcn_wmma_f32_16x16x32_bf16(
                    false, ah.v, false, bl[n].v, (short)0, acc[t][n], false, false);
                acc[t][n] = __builtin_amdgcn_wmma_f32_16x16x32_bf16(
                    false, al.v, false, bh[n].v, (short)0, acc[t][n], false, false);
            }
        }
        __syncthreads();
        buf ^= 1;
    }

    // Epilogue: logit[b,s] += sum_m V[m] * tanh(u1[b,m] + bias[m] + u2[m,s]).
    // C/D layout: VGPR j -> M = hl*8 + j (within tile), N = l15.
    const float* u1row = u1 + (size_t)b * H_;
    float part0 = 0.0f, part1 = 0.0f;
#pragma unroll
    for (int t = 0; t < 4; ++t) {
#pragma unroll
        for (int j = 0; j < 8; ++j) {
            int m = mbase + t * 16 + hl * 8 + j;
            float vm = Vv[m];
            float um = u1row[m] + Wref_b[m];
            part0 += vm * tanhf(um + acc[t][0][j]);
            part1 += vm * tanhf(um + acc[t][1][j]);
        }
    }
    atomicAdd(&red[l15], part0);
    atomicAdd(&red[16 + l15], part1);
    __syncthreads();
    if (tid < 32) logit[(size_t)b * S_ + s0 + tid] = red[tid];
}

// ---------------------------------------------------------------------------
// Softmax over S per batch row.
// ---------------------------------------------------------------------------
__global__ __launch_bounds__(256)
void k_softmax(const float* __restrict__ logit, float* __restrict__ probs) {
    int b = blockIdx.x, tid = threadIdx.x;
    __shared__ float sred[256];
    float l[8], mx = -3.4e38f;
#pragma unroll
    for (int i = 0; i < 8; ++i) {
        l[i] = logit[(size_t)b * S_ + tid + i * 256];
        mx = fmaxf(mx, l[i]);
    }
    sred[tid] = mx;
    __syncthreads();
    for (int off = 128; off; off >>= 1) {
        if (tid < off) sred[tid] = fmaxf(sred[tid], sred[tid + off]);
        __syncthreads();
    }
    mx = sred[0];
    __syncthreads();
    float e[8], sum = 0.0f;
#pragma unroll
    for (int i = 0; i < 8; ++i) { e[i] = expf(l[i] - mx); sum += e[i]; }
    sred[tid] = sum;
    __syncthreads();
    for (int off = 128; off; off >>= 1) {
        if (tid < off) sred[tid] += sred[tid + off];
        __syncthreads();
    }
    float inv = 1.0f / sred[0];
#pragma unroll
    for (int i = 0; i < 8; ++i)
        probs[(size_t)b * S_ + tid + i * 256] = e[i] * inv;
}

__global__ __launch_bounds__(256)
void k_zero(float* __restrict__ p) {
    p[blockIdx.x * 256 + threadIdx.x] = 0.0f;
}

// ---------------------------------------------------------------------------
// wsum[b,h] = sum_s probs[b,s] * src[b,s,h]   (d' = Wref @ wsum + bref)
// ---------------------------------------------------------------------------
__global__ __launch_bounds__(256)
void k_wsum(const float* __restrict__ src, const float* __restrict__ probs,
            float* __restrict__ wsum) {
    int b = blockIdx.y, chunk = blockIdx.x, tid = threadIdx.x;
    __shared__ float sp[256];
    int sbase = chunk * 256;
    sp[tid] = probs[(size_t)b * S_ + sbase + tid];
    __syncthreads();
    float a0 = 0.0f, a1 = 0.0f;
    const float* base = src + ((size_t)b * S_ + sbase) * H_;
    for (int s = 0; s < 256; ++s) {
        float p = sp[s];
        a0 += p * base[(size_t)s * H_ + tid];
        a1 += p * base[(size_t)s * H_ + tid + 256];
    }
    atomicAdd(&wsum[(size_t)b * H_ + tid], a0);
    atomicAdd(&wsum[(size_t)b * H_ + tid + 256], a1);
}

// ---------------------------------------------------------------------------
extern "C" void kernel_launch(void* const* d_in, const int* in_sizes, int n_in,
                              void* d_out, int out_size, void* d_ws, size_t ws_size,
                              hipStream_t stream) {
    const float* src    = (const float*)d_in[0];
    const float* tgt    = (const float*)d_in[1];
    const float* Wq_w   = (const float*)d_in[2];
    const float* Wq_b   = (const float*)d_in[3];
    const float* Wref_w = (const float*)d_in[4];
    const float* Wref_b = (const float*)d_in[5];
    const float* Vv     = (const float*)d_in[6];

    float* out_dprime = (float*)d_out;            // (B,H)
    float* out_probs  = out_dprime + B_ * H_;     // (B,S)
    float* out_logit  = out_probs + B_ * S_;      // (B,S)

    float*  u1    = (float*)d_ws;                 // B*H
    float*  wsum  = u1 + B_ * H_;                 // B*H
    __bf16* Whi   = (__bf16*)(wsum + B_ * H_);    // H*H
    __bf16* Wlo   = Whi + (size_t)H_ * H_;        // H*H

    // 1. Split Wref_w into bf16 hi/lo.
    k_convert<<<(H_ * H_) / 256, 256, 0, stream>>>(Wref_w, Whi, Wlo);
    // 2. u1 = tgt @ Wq_w^T + Wq_b.
    k_gemv<<<(B_ * H_) / 8, 256, 0, stream>>>(Wq_w, tgt, Wq_b, u1);
    // 3. Fused WMMA GEMM + tanh/V epilogue -> logits.
    k_main<<<dim3(S_ / 32, B_), 256, 0, stream>>>(src, Whi, Wlo, Wref_b, Vv,
                                                  u1, out_logit);
    // 4. Softmax -> probs.
    k_softmax<<<B_, 256, 0, stream>>>(out_logit, out_probs);
    // 5. wsum = src^T @ probs (per batch).
    k_zero<<<(B_ * H_) / 256, 256, 0, stream>>>(wsum);
    k_wsum<<<dim3(8, B_), 256, 0, stream>>>(src, out_probs, wsum);
    // 6. d' = Wref_w @ wsum + Wref_b.
    k_gemv<<<(B_ * H_) / 8, 256, 0, stream>>>(Wref_w, wsum, Wref_b, out_dprime);
}
